// CAPBlock_57105885167636
// MI455X (gfx1250) — matrix operand order repaired and checked by reading, not compiled, not run on hardware
//
#include <hip/hip_runtime.h>
#include <stdint.h>

// ---------------------------------------------------------------------------
// CDNA5 (gfx1250) fused DBA-L1 attention
//   proj_wmma:  q/k/v = X @ W^T + b   (f16 WMMA, f32 accum) -> f16 in ws
//   attn_l1_fused: streaming L1-distance logits + rel-softmax + P@V WMMA,
//                  V chunk staged into LDS by the Tensor Data Mover (TDM).
// Problem constants (from reference): B=4, Q=256, K=4096, E=512, H=8, Dh=64
// ---------------------------------------------------------------------------

typedef __attribute__((ext_vector_type(16))) _Float16     v16h;
typedef __attribute__((ext_vector_type(8)))  float        v8f;
typedef __attribute__((ext_vector_type(8)))  _Float16     h8;
typedef __attribute__((ext_vector_type(4)))  unsigned int u32x4;
typedef __attribute__((ext_vector_type(4)))  int          i32x4;
typedef __attribute__((ext_vector_type(8)))  int          i32x8;

#define EIN   512
#define EOUT  512
#define NB    4
#define NQ    256
#define NK    4096
#define NH    8
#define DHD   64
#define KT    64      // K-chunk per iteration in attention kernel

#define L1_SCALING      6.8200176f          // sqrt(0.72676) * sqrt(64)
#define L1_CENTER       72.216320f          // 64 * 1.12838
#define INV_L1_SCALING  (1.0f / L1_SCALING)

// ---------------------------------------------------------------------------
// Kernel 1: out[r, n] = (f16) ( sum_e X[r,e] * W[n,e] + bias[n] )
// grid = (rows/32, 512/64), block = 256 (8 waves); each wave owns one 16x16
// C tile of the 32x64 block tile; K-loop steps of 32 using wmma f32_16x16x32_f16.
// ---------------------------------------------------------------------------
__global__ __launch_bounds__(256)
void proj_wmma(const float* __restrict__ X, const float* __restrict__ W,
               const float* __restrict__ bias, _Float16* __restrict__ out) {
    __shared__ __align__(16) _Float16 As[32][40];   // 32 rows x 32 k (padded)
    __shared__ __align__(16) _Float16 Bs[64][40];   // 64 cols x 32 k (padded)

    const int t    = threadIdx.x;
    const int lane = t & 31;
    const int wave = t >> 5;
    const int mt   = wave & 1;          // 2 M tiles
    const int nt   = wave >> 1;         // 4 N tiles
    const size_t row0 = (size_t)blockIdx.x * 32;
    const int    col0 = blockIdx.y * 64;

    v8f acc = {};

    for (int ke = 0; ke < EIN; ke += 32) {
        { // A tile: 32x32 f32 -> f16 LDS (one float4 per thread)
            const int r = t >> 3, c = (t & 7) * 4;
            const float4 v = *(const float4*)(X + (row0 + r) * EIN + ke + c);
            As[r][c]     = (_Float16)v.x;
            As[r][c + 1] = (_Float16)v.y;
            As[r][c + 2] = (_Float16)v.z;
            As[r][c + 3] = (_Float16)v.w;
        }
        #pragma unroll
        for (int i = 0; i < 2; ++i) { // B tile: W[col, ke..ke+31] (two float4/thread)
            const int lin = t + i * 256;
            const int n = lin >> 3, c = (lin & 7) * 4;
            const float4 v = *(const float4*)(W + (size_t)(col0 + n) * EIN + ke + c);
            Bs[n][c]     = (_Float16)v.x;
            Bs[n][c + 1] = (_Float16)v.y;
            Bs[n][c + 2] = (_Float16)v.z;
            Bs[n][c + 3] = (_Float16)v.w;
        }
        __syncthreads();

        // Fragments in contiguous k order (k = 16*(lane/16) + s), same for A and B.
        v16h a, bf;
        const int ar  = mt * 16 + (lane & 15);
        const int akb = (lane >> 4) * 16;
        #pragma unroll
        for (int i = 0; i < 8; ++i) {
            a[i]     = As[ar][akb + i];
            a[i + 8] = As[ar][akb + 8 + i];
        }
        const int bn  = nt * 16 + (lane & 15);
        const int bkb = (lane >> 4) * 16;
        #pragma unroll
        for (int i = 0; i < 16; ++i) bf[i] = Bs[bn][bkb + i];

        acc = __builtin_amdgcn_wmma_f32_16x16x32_f16(false, a, false, bf,
                                                     (short)0, acc, false, false);
        __syncthreads();
    }

    const int   ncol  = col0 + nt * 16 + (lane & 15);
    const float bb    = bias[ncol];
    const size_t rbase = row0 + mt * 16 + (size_t)((lane >> 4) * 8);
    #pragma unroll
    for (int r = 0; r < 8; ++r)
        out[(rbase + r) * EOUT + ncol] = (_Float16)(acc[r] + bb);
}

// ---------------------------------------------------------------------------
// Kernel 2: fused L1-logits + rel-normalize + P@V.
// One block per (b, h, 32-query tile); 256 threads = 8 waves.
// V chunk is DMA'd by the TDM (tensor_load_to_lds) with LDS row padding
// (32-DWORD rows + 4-DWORD pad == vs[64][72] layout); K chunk is loaded
// manually because it needs f16->f32 promotion for the L1 core.
// ---------------------------------------------------------------------------
__global__ __launch_bounds__(256)
void attn_l1_fused(const _Float16* __restrict__ qh, const _Float16* __restrict__ kh,
                   const _Float16* __restrict__ vh, const unsigned char* __restrict__ mask,
                   const float* __restrict__ rel, const float* __restrict__ adiag,
                   float* __restrict__ out) {
    __shared__ __align__(16) float    qs[32][68];   // q tile, f32, padded rows
    __shared__ __align__(16) float    ks[KT][68];   // k chunk, f32, padded rows
    __shared__ __align__(16) _Float16 vs[KT][72];   // v chunk, f16 (TDM destination)
    __shared__ __align__(16) _Float16 Ps[32][72];   // sfmx weights, f16 (WMMA A)
    __shared__ __align__(16) float    dg[DHD];
    __shared__ float msk[KT];
    __shared__ float rowsum[32];

    const int t = threadIdx.x, lane = t & 31, wave = t >> 5;
    const int qt = blockIdx.x & 7;          // 8 query tiles of 32
    const int bh = blockIdx.x >> 3;
    const int h  = bh & (NH - 1);
    const int b  = bh >> 3;

    { // q tile: 32 rows x 64 halfs -> f32 LDS
        const int r = t >> 3, c = (t & 7) * 8;
        const h8 v = *(const h8*)(qh + ((size_t)(b * NQ + qt * 32 + r)) * EOUT + h * DHD + c);
        #pragma unroll
        for (int i = 0; i < 8; ++i) qs[r][c + i] = (float)v[i];
    }
    if (t < DHD) dg[t] = adiag[h * DHD + t];
    if (t < 32)  rowsum[t] = 0.f;
    const float adenom = rel[b * NH + h];

    const int qi = t >> 3;                  // each thread owns one query row
    const int k0 = t & 7;                   // ... and keys {k0 + 8j}, j=0..7
    const int mt = wave & 1, nt = wave >> 1;
    float rs_part = 0.f;
    v8f acc = {};

    for (int kc = 0; kc < NK; kc += KT) {
        const size_t chunk = ((size_t)(b * NK + kc)) * EOUT + h * DHD;

#if __has_builtin(__builtin_amdgcn_tensor_load_to_lds)
        if (wave == 0) { // one DMA per workgroup (TDM ignores EXEC; issue per-wave)
            const uint64_t gaddr  = (uint64_t)(uintptr_t)(vh + chunk);
            const uint32_t ldsoff = (uint32_t)(uintptr_t)&vs[0][0];
            // D# group0: count=1 | lds_addr | global_addr[56:0] | type=2
            u32x4 g0 = { 1u, ldsoff, (unsigned int)gaddr,
                         (unsigned int)((gaddr >> 32) & 0x01FFFFFFu) | (2u << 30) };
            // D# group1: data_size=2B, pad_enable, pad every 32 DWORDs by 4 DWORDs,
            //            tensor_dim0=64, tensor_dim1=64, tile 64x64, row stride 512.
            i32x8 g1 = { (int)((1u << 16) | (1u << 20) | (4u << 22) | (3u << 25)),
                         (int)(64u << 16),   // tensor_dim0[15:0]
                         (int)(64u << 16),   // tensor_dim1[15:0]
                         (int)(64u << 16),   // tile_dim0
                         (int)64,            // tile_dim1 (tile_dim2 = 0)
                         (int)512,           // tensor_dim0_stride[31:0]
                         0, 0 };
            i32x4 gz = { 0, 0, 0, 0 };
#if __clang_major__ >= 23
            i32x8 gz8 = { 0, 0, 0, 0, 0, 0, 0, 0 };
            __builtin_amdgcn_tensor_load_to_lds(g0, g1, gz, gz, gz8, 0);
#else
            __builtin_amdgcn_tensor_load_to_lds(g0, g1, gz, gz, 0);
#endif
        }
#endif
        { // k chunk: 64 rows x 64 halfs -> f32 LDS (two h8 loads per thread)
            const int r = t >> 2, c0 = (t & 3) * 16;
            const size_t gbase = chunk + (size_t)r * EOUT;
            #pragma unroll
            for (int j = 0; j < 2; ++j) {
                const int c = c0 + j * 8;
                const h8 kv = *(const h8*)(kh + gbase + c);
                #pragma unroll
                for (int i = 0; i < 8; ++i) ks[r][c + i] = (float)kv[i];
            }
            if (kc + KT < NK) // gfx1250 global_prefetch_b8 for next chunk
                __builtin_prefetch(kh + gbase + (size_t)KT * EOUT, 0, 0);
        }
        if (t < KT) msk[t] = mask[(size_t)b * NK + kc + t] ? 1.f : 0.f;

#if __has_builtin(__builtin_amdgcn_tensor_load_to_lds)
        if (wave == 0) __builtin_amdgcn_s_wait_tensorcnt(0);
#else
        { // fallback: manual V chunk load
            const int r = t >> 2, c0 = (t & 3) * 16;
            const size_t gbase = chunk + (size_t)r * EOUT;
            #pragma unroll
            for (int j = 0; j < 2; ++j) {
                const int c = c0 + j * 8;
                *(h8*)&vs[r][c] = *(const h8*)(vh + gbase + c);
            }
        }
#endif
        __syncthreads();

        // weighted-L1 logits: 8 key accumulators per thread; qv/dv hoisted so the
        // hot loop is one ds_load_b128 (consecutive ks rows -> conflict-free) + fmas.
        float sacc[8] = {0.f, 0.f, 0.f, 0.f, 0.f, 0.f, 0.f, 0.f};
        #pragma unroll 4
        for (int d = 0; d < DHD; d += 4) {
            const float4 qv = *(const float4*)&qs[qi][d];
            const float4 dv = *(const float4*)&dg[d];
            #pragma unroll
            for (int j = 0; j < 8; ++j) {
                const float4 kv = *(const float4*)&ks[k0 + 8 * j][d];
                sacc[j] = fmaf(dv.x, fabsf(kv.x - qv.x), sacc[j]);
                sacc[j] = fmaf(dv.y, fabsf(kv.y - qv.y), sacc[j]);
                sacc[j] = fmaf(dv.z, fabsf(kv.z - qv.z), sacc[j]);
                sacc[j] = fmaf(dv.w, fabsf(kv.w - qv.w), sacc[j]);
            }
        }
        #pragma unroll
        for (int j = 0; j < 8; ++j) {
            const int kk = k0 + 8 * j;
            const float logit = (L1_CENTER - sacc[j]) * INV_L1_SCALING;
            const float ae = msk[kk] * __expf(logit);  // mask==0 -> exact 0 (NEG_INF path)
            const float sf = ae / (ae + adenom);
            rs_part += sf;
            Ps[qi][kk] = (_Float16)sf;
        }
        __syncthreads();

        // P(32x64) @ V(64x64) via two 16x16x32 WMMA k-steps per wave tile
        #pragma unroll
        for (int k2 = 0; k2 < 2; ++k2) {
            v16h a, bf;
            const int ar  = mt * 16 + (lane & 15);
            const int akb = k2 * 32 + (lane >> 4) * 16;
            #pragma unroll
            for (int i = 0; i < 8; ++i) {
                a[i]     = Ps[ar][akb + i];
                a[i + 8] = Ps[ar][akb + 8 + i];
            }
            const int bn  = nt * 16 + (lane & 15);
            const int bkb = k2 * 32 + (lane >> 4) * 16;
            #pragma unroll
            for (int i = 0; i < 16; ++i) bf[i] = vs[bkb + i][bn];
            acc = __builtin_amdgcn_wmma_f32_16x16x32_f16(false, a, false, bf,
                                                         (short)0, acc, false, false);
        }
        __syncthreads();
    }

    atomicAdd(&rowsum[qi], rs_part);   // ds_add_f32 reduction (8 partials per row)
    __syncthreads();

    // out = acc / max(rowsum, 1e-12); C/D layout: lane -> col, VGPR r -> row.
    const int ncol = nt * 16 + (lane & 15);
    const int rb   = mt * 16 + (lane >> 4) * 8;
    #pragma unroll
    for (int r = 0; r < 8; ++r) {
        const int qr  = rb + r;
        const float den = fmaxf(rowsum[qr], 1e-12f);
        out[((size_t)(b * NQ + qt * 32 + qr)) * EOUT + h * DHD + ncol] = acc[r] / den;
    }
}

// ---------------------------------------------------------------------------
// Host-side launch. ws layout (33 MB, fits easily in L2 for k/v re-reads):
//   [0,      1 MB) q_f16  [1024  x 512]
//   [1 MB,  17 MB) k_f16  [16384 x 512]
//   [17 MB, 33 MB) v_f16  [16384 x 512]
// ---------------------------------------------------------------------------
extern "C" void kernel_launch(void* const* d_in, const int* in_sizes, int n_in,
                              void* d_out, int out_size, void* d_ws, size_t ws_size,
                              hipStream_t stream) {
    (void)in_sizes; (void)n_in; (void)out_size; (void)ws_size;
    const float*         qin  = (const float*)d_in[0];         // [4,256,512]
    const float*         bag  = (const float*)d_in[1];         // [4,4096,512]
    const unsigned char* bmsk = (const unsigned char*)d_in[2]; // [4,4096] bool
    const float*         rel  = (const float*)d_in[3];         // [4,8,1,1]
    const float*         Wqk  = (const float*)d_in[4];         // [512,512]
    const float*         bqk  = (const float*)d_in[5];         // [512]
    const float*         Wv   = (const float*)d_in[6];         // [512,512]
    const float*         bv   = (const float*)d_in[7];         // [512]
    const float*         adia = (const float*)d_in[8];         // [8,1,64]
    float*               outp = (float*)d_out;                 // [4,256,512]

    char* ws = (char*)d_ws;
    _Float16* qh = (_Float16*)(ws);
    _Float16* kh = (_Float16*)(ws + ((size_t)1 << 20));
    _Float16* vh = (_Float16*)(ws + ((size_t)17 << 20));

    const dim3 blk(256);
    proj_wmma<<<dim3(NB * NQ / 32, EOUT / 64), blk, 0, stream>>>(qin, Wqk, bqk, qh);
    proj_wmma<<<dim3(NB * NK / 32, EOUT / 64), blk, 0, stream>>>(bag, Wqk, bqk, kh);
    proj_wmma<<<dim3(NB * NK / 32, EOUT / 64), blk, 0, stream>>>(bag, Wv, bv, vh);
    attn_l1_fused<<<dim3(NB * NH * (NQ / 32)), blk, 0, stream>>>(qh, kh, vh, bmsk,
                                                                 rel, adia, outp);
}
